// MultiheadEnergyAttention_40054865002641
// MI455X (gfx1250) — matrix operand order repaired and checked by reading, not compiled, not run on hardware
//
#include <hip/hip_runtime.h>
#include <math.h>

typedef __attribute__((ext_vector_type(16))) _Float16 v16h;
typedef __attribute__((ext_vector_type(8)))  float    v8f;

#define EMBED  768
#define HEADS  12
#define HDIM   64
#define BATCHN 8
#define SEQL   1024
#define MROWS  (BATCHN * SEQL)                 // 8192
#define NTILES (BATCHN * HEADS * (SEQL / 16))  // 6144 energy tiles

#define M_INIT   (-1e30f)   // online-max sentinel (finite: no inf-inf NaN paths)
#define M_MASKED (-2e30f)   // masked-score sentinel: exp(M_MASKED - m) == 0 always

// ---------------------------------------------------------------------------
// Stage 1: projection GEMM.  Out[m,n] = sum_k X[m,k] * W[n,k]  (fp16 output)
//   z==0 computes Q and folds beta_h into it (scores = (beta*q) . k), so the
//   energy kernel needs no per-score multiply. z==1 computes K unscaled.
//   One wave owns one 16x16 tile; K loop in steps of 32 via
//   v_wmma_f32_16x16x32_f16 with f32 accumulation.
// ---------------------------------------------------------------------------
__global__ __launch_bounds__(128)
void proj16_kernel(const float* __restrict__ x,
                   const float* __restrict__ wq,
                   const float* __restrict__ wk,
                   const float* __restrict__ beta,
                   _Float16* __restrict__ qout,
                   _Float16* __restrict__ kout)
{
    const int lane = threadIdx.x & 31;
    const int wav  = threadIdx.x >> 5;
    const int r  = lane & 15;   // row (A) / column (B) / N (C)
    const int hx = lane >> 4;   // half-wave select

    const int m0 = (blockIdx.x * 4 + wav) * 16;
    const int n0 = blockIdx.y * 16;
    const float* w   = (blockIdx.z == 0) ? wq : wk;
    _Float16*    out = (blockIdx.z == 0) ? qout : kout;
    // 16-col tile lies inside one head (16 | 64): uniform scale per tile
    const float scale = (blockIdx.z == 0) ? beta[n0 >> 6] : 1.0f;

    const float* arow = x + (size_t)(m0 + r) * EMBED;
    const float* brow = w + (size_t)(n0 + r) * EMBED;

    v8f c = {};
    #pragma unroll 4
    for (int k0 = 0; k0 < EMBED; k0 += 32) {
        // A fragment (16x32 f16): lane (hx,r) holds row r, K = hx*8+e, 16+hx*8+e
        v16h A;
        const float* pa = arow + k0;
        #pragma unroll
        for (int e = 0; e < 8; ++e) A[e]     = (_Float16)pa[hx * 8 + e];
        #pragma unroll
        for (int e = 0; e < 8; ++e) A[8 + e] = (_Float16)pa[16 + hx * 8 + e];
        // B fragment (32x16 f16): lane holds column n0+r, K = hx*8+e, 16+hx*8+e
        v16h B;
        const float* pb = brow + k0;
        #pragma unroll
        for (int e = 0; e < 8; ++e) { B[e] = (_Float16)pb[hx * 8 + e]; B[8 + e] = (_Float16)pb[16 + hx * 8 + e]; }

        c = __builtin_amdgcn_wmma_f32_16x16x32_f16(false, A, false, B,
                                                   (short)0, c, false, false);
    }
    // C layout: element v -> (M = v + 8*hx, N = r)
    #pragma unroll
    for (int v = 0; v < 8; ++v) {
        const int m = m0 + v + 8 * hx;
        out[(size_t)m * EMBED + n0 + r] = (_Float16)(c[v] * scale);
    }
}

// ---------------------------------------------------------------------------
// Stage 2: fused scores + masked online logsumexp + energy partials.
//   One wave owns (b, h, 16 query rows); streams all 64 key tiles, 2 WMMAs
//   each. C layout puts score[srow][t] with t = tt*16 + (lane&15): each lane
//   sees a fixed column residue class, so it keeps a private per-row online
//   (max, sumexp) over its 64 columns -- NO shuffles and NO EXEC divergence
//   in the hot loop (diagonal mask = branchless finite-sentinel cndmask; the
//   tt==st predicate half is scalar-uniform). The 16-lane log-domain merge
//   runs once at the end (4 shfl_xor steps).
// ---------------------------------------------------------------------------
__global__ __launch_bounds__(128)
void energy_kernel(const _Float16* __restrict__ Q,
                   const _Float16* __restrict__ K,
                   const float* __restrict__ beta,
                   float* __restrict__ partials)
{
    const int lane = threadIdx.x & 31;
    const int wav  = threadIdx.x >> 5;
    const int r  = lane & 15;
    const int hx = lane >> 4;

    const int tile = blockIdx.x * 4 + wav;      // 0 .. NTILES-1
    const int st = tile & 63;                   // s tile (16 rows)
    const int h  = (tile >> 6) % HEADS;
    const int b  = tile / (64 * HEADS);

    // A fragments for the two k-steps of head dim 64 (reused for all key tiles)
    const _Float16* qrow = Q + (size_t)(b * SEQL + st * 16 + r) * EMBED + h * HDIM;
    v16h A0, A1;
    #pragma unroll
    for (int e = 0; e < 8; ++e) { A0[e] = qrow[hx*8 + e];      A0[8+e] = qrow[16 + hx*8 + e]; }
    #pragma unroll
    for (int e = 0; e < 8; ++e) { A1[e] = qrow[32 + hx*8 + e]; A1[8+e] = qrow[48 + hx*8 + e]; }

    // per-lane diagonal predicate: this lane's column residue hits row v+8*hx
    bool dlane[8];
    #pragma unroll
    for (int v = 0; v < 8; ++v) dlane[v] = (v + 8 * hx) == r;

    // per-lane online LSE state over this lane's column subset
    float m_run[8], l_run[8];
    #pragma unroll
    for (int v = 0; v < 8; ++v) { m_run[v] = M_INIT; l_run[v] = 0.0f; }

    for (int tt = 0; tt < SEQL / 16; ++tt) {
        const _Float16* krow = K + (size_t)(b * SEQL + tt * 16 + r) * EMBED + h * HDIM;
        v16h B0, B1;
        #pragma unroll
        for (int e = 0; e < 8; ++e) { B0[e] = krow[hx * 8 + e]; B0[8 + e] = krow[16 + hx * 8 + e]; }
        #pragma unroll
        for (int e = 0; e < 8; ++e) { B1[e] = krow[32 + hx * 8 + e]; B1[8 + e] = krow[48 + hx * 8 + e]; }

        v8f c = {};
        c = __builtin_amdgcn_wmma_f32_16x16x32_f16(false, A0, false, B0, (short)0, c, false, false);
        c = __builtin_amdgcn_wmma_f32_16x16x32_f16(false, A1, false, B1, (short)0, c, false, false);

        const bool dtile = (tt == st);          // uniform: diagonal only in matching block
        #pragma unroll
        for (int v = 0; v < 8; ++v) {
            // beta already folded into Q; branchless diagonal mask via sentinel:
            // exp(M_MASKED - nm) underflows to exactly 0 (all-finite arithmetic)
            const float val = (dtile && dlane[v]) ? M_MASKED : c[v];
            const float nm  = fmaxf(m_run[v], val);
            l_run[v] = l_run[v] * __expf(m_run[v] - nm) + __expf(val - nm);
            m_run[v] = nm;
        }
    }

    // merge the 16 per-lane (m,l) pairs per row in log domain, then lse = m+log(l)
    float part = 0.0f;
    #pragma unroll
    for (int v = 0; v < 8; ++v) {
        float m = m_run[v], l = l_run[v];
        #pragma unroll
        for (int d = 1; d <= 8; d <<= 1) {
            const float mo = __shfl_xor(m, d, 16);
            const float lo = __shfl_xor(l, d, 16);
            const float nm = fmaxf(m, mo);
            l = l * __expf(m - nm) + lo * __expf(mo - nm);
            m = nm;
        }
        part += m + __logf(l);
    }
    // lanes 0-15 summed rows 0-7, lanes 16-31 rows 8-15: combine halves
    part += __shfl_xor(part, 16, 32);
    if (lane == 0) partials[tile] = -part / beta[h];
}

// ---------------------------------------------------------------------------
// Stage 3: deterministic scalar reduction of the 6144 tile partials.
// ---------------------------------------------------------------------------
__global__ __launch_bounds__(256)
void reduce_kernel(const float* __restrict__ partials, int n, float* __restrict__ out)
{
    __shared__ float sm[256];
    float s = 0.0f;
    for (int i = threadIdx.x; i < n; i += 256) s += partials[i];
    sm[threadIdx.x] = s;
    __syncthreads();
    for (int w = 128; w > 0; w >>= 1) {
        if ((int)threadIdx.x < w) sm[threadIdx.x] += sm[threadIdx.x + w];
        __syncthreads();
    }
    if (threadIdx.x == 0) *out = sm[0];
}

extern "C" void kernel_launch(void* const* d_in, const int* in_sizes, int n_in,
                              void* d_out, int out_size, void* d_ws, size_t ws_size,
                              hipStream_t stream) {
    const float* x    = (const float*)d_in[0];
    const float* wq   = (const float*)d_in[1];
    const float* wk   = (const float*)d_in[2];
    const float* beta = (const float*)d_in[3];
    float* out = (float*)d_out;

    // workspace layout: Q(beta-scaled) fp16 (12 MB) | K fp16 (12 MB) | partials (24 KB)
    _Float16* Q  = (_Float16*)d_ws;
    _Float16* Kh = Q + (size_t)MROWS * EMBED;
    float* partials = (float*)(Kh + (size_t)MROWS * EMBED);

    // Stage 1: Q (beta-folded) and K projections, 4 waves/block
    dim3 g1(MROWS / 64, EMBED / 16, 2);
    proj16_kernel<<<g1, 128, 0, stream>>>(x, wq, wk, beta, Q, Kh);

    // Stage 2: fused scores + LSE, one wave per (b, h, 16-row s-tile)
    dim3 g2(NTILES / 4);
    energy_kernel<<<g2, 128, 0, stream>>>(Q, Kh, beta, partials);

    // Stage 3: deterministic final reduction to the scalar energy
    reduce_kernel<<<1, 256, 0, stream>>>(partials, NTILES, out);
}
